// CRDLoss_15685220565755
// MI455X (gfx1250) — matrix operand order repaired and checked by reading, not compiled, not run on hardware
//
#include <hip/hip_runtime.h>
#include <math.h>

typedef __attribute__((ext_vector_type(16))) _Float16 v16h;
typedef __attribute__((ext_vector_type(8)))  float    v8f;

#define B_SZ        256
#define K_NEG       4096
#define KP1         4097
#define FD          128
#define SDIM        2048
#define NDATA       1000000
#define NPE         (B_SZ * KP1)        /* 1,048,832 */
#define TILES_PER_B 257                 /* ceil(4097/16) */
#define MAIN_BLOCKS 512
#define MAIN_THREADS 256
#define WAVES_PER_BLOCK (MAIN_THREADS / 32)
#define TOTAL_WAVES (MAIN_BLOCKS * WAVES_PER_BLOCK) /* 4096 = 512 (dir,b) x 8 */
#define SUBWAVES    8                   /* waves cooperating on one (dir,b) */
#define INV_T       (1.0f / 0.07f)
#define RES_C       0.004096f           /* K/N_DATA */
#define EPS_C       1e-7f

// ---------------------------------------------------------------------------
// map[i] = -1 everywhere, then map[idx[j]] = j : redirect gathers that hit the
// 256 momentum-updated memory rows to the freshly computed rows in `upd*`.
// ---------------------------------------------------------------------------
__global__ void k_init_map(int* __restrict__ map) {
  for (int i = blockIdx.x * blockDim.x + threadIdx.x; i < NDATA;
       i += gridDim.x * blockDim.x)
    map[i] = -1;
}

__global__ void k_scatter(const int* __restrict__ idx, int* __restrict__ map) {
  const int t = threadIdx.x;
  if (t < B_SZ) map[idx[t]] = t;
}

// ---------------------------------------------------------------------------
// Projection: e = normalize(f @ W^T + b), plus momentum update of memory rows:
// upd = normalize(0.5*mem[idx[b]] + 0.5*e).  One block per (b, direction).
// Tiny op (0.27 GFLOP total); W (1 MB) stays L2-resident across blocks.
// ---------------------------------------------------------------------------
__global__ void __launch_bounds__(128) k_proj(
    const float* __restrict__ f_s, const float* __restrict__ f_t,
    const float* __restrict__ W_s, const float* __restrict__ b_s,
    const float* __restrict__ W_t, const float* __restrict__ b_t,
    const float* __restrict__ mem1, const float* __restrict__ mem2,
    const int* __restrict__ idx,
    float* __restrict__ es, float* __restrict__ et,
    float* __restrict__ upd1, float* __restrict__ upd2)
{
  __shared__ __attribute__((aligned(16))) float fbuf[SDIM];
  __shared__ float red[128];
  const int b   = blockIdx.x;
  const int dir = blockIdx.y;
  const int t   = threadIdx.x;

  const float* f    = (dir ? f_t : f_s) + (size_t)b * SDIM;
  const float* W    = (dir ? W_t : W_s);
  const float* bias = (dir ? b_t : b_s);
  const float* mem  = (dir ? mem2 : mem1);
  float* e_out = (dir ? et : es);
  float* u_out = (dir ? upd2 : upd1);

  for (int i = t; i < SDIM / 4; i += 128)
    ((float4*)fbuf)[i] = ((const float4*)f)[i];
  __syncthreads();

  float acc = bias[t];
  const float* wr = W + (size_t)t * SDIM;
  for (int k = 0; k < SDIM / 4; ++k) {
    const float4 w = ((const float4*)wr)[k];
    const float4 x = ((const float4*)fbuf)[k];
    acc += w.x * x.x + w.y * x.y + w.z * x.z + w.w * x.w;
  }

  red[t] = acc * acc; __syncthreads();
  for (int s = 64; s > 0; s >>= 1) { if (t < s) red[t] += red[t + s]; __syncthreads(); }
  const float nrm = fmaxf(sqrtf(red[0]), 1e-12f);
  __syncthreads();
  const float e = acc / nrm;
  e_out[b * FD + t] = e;

  const float mv = mem[(size_t)idx[b] * FD + t];
  const float u = 0.5f * mv + 0.5f * e;
  red[t] = u * u; __syncthreads();
  for (int s = 64; s > 0; s >>= 1) { if (t < s) red[t] += red[t + s]; __syncthreads(); }
  const float un = sqrtf(red[0]);
  u_out[b * FD + t] = u / un;
}

// ---------------------------------------------------------------------------
// Main HBM-bound pass.  Wave->work map: wid in [0,4096); pair = wid/8 selects
// (dir, b) -- so the B operand (e[b] replicated into all 16 WMMA columns),
// the mem/upd pointers and the output row base are loop-invariant and hoisted.
// sub = wid%8 strides over the 257 row-tiles of that (dir, b).
// Per tile: gather 16 rows (512 B each, random rows of the 512 MB bank),
// 4x v_wmma_f32_16x16x32_f16 accumulate the K=128 dots, then exp(dot/T).
// VGPR layouts per CDNA5 ISA 7.12.2 (wave32):
//   A 16x32 f16 : lane L holds row m=L%16; elems 0-7 = K 8h..8h+7,
//                 elems 8-15 = K 16+8h..16+8h+7, h=L/16.
//   B 32x16 f16 : lane L = column n=L%16; same K order as A.
//   D 16x16 f32 : lane L col n=L%16; VGPR v = row v+8h.
// Per-wave Z partials go to fixed slots (deterministic reduction later).
// ---------------------------------------------------------------------------
__global__ void __launch_bounds__(MAIN_THREADS) k_main(
    const float* __restrict__ es,   const float* __restrict__ et,
    const float* __restrict__ upd1, const float* __restrict__ upd2,
    const float* __restrict__ mem1, const float* __restrict__ mem2,
    const int* __restrict__ idx,    const int* __restrict__ cidx,
    const int* __restrict__ map,
    float* __restrict__ raw, float* __restrict__ zslot)
{
  const int lane = threadIdx.x & 31;
  const int wid  = blockIdx.x * WAVES_PER_BLOCK + (threadIdx.x >> 5);
  const int m = lane & 15;     // matrix row this lane owns (A) / column (B,D)
  const int h = lane >> 4;     // K-half selector

  const int pair = wid >> 3;            // 0..511
  const int sub  = wid & (SUBWAVES - 1);
  const int dir  = pair >> 8;           // 0: out_s (mem2 . es), 1: out_t
  const int b    = pair & (B_SZ - 1);

  const float* evec = (dir == 0 ? es : et) + b * FD;
  const float* mem  = (dir == 0 ? mem2 : mem1);
  const float* upd  = (dir == 0 ? upd2 : upd1);
  float* rawp = raw + (size_t)dir * NPE + (size_t)b * KP1;
  const int pos_ridx = idx[b];          // row index of the positive sample
  const int* crow = cidx + b * K_NEG;

  // B operand (loop-invariant): 4 K-chunks of e[b], replicated over 16 cols.
  v16h bv[4];
#pragma unroll
  for (int c = 0; c < 4; ++c) {
    float tb[16] __attribute__((aligned(16)));
    const float4* pb = (const float4*)(evec + 32 * c + 8 * h);
    *(float4*)(tb + 0)  = pb[0];
    *(float4*)(tb + 4)  = pb[1];
    *(float4*)(tb + 8)  = pb[4];
    *(float4*)(tb + 12) = pb[5];
#pragma unroll
    for (int q = 0; q < 16; ++q) bv[c][q] = (_Float16)tb[q];
  }

  float zs = 0.f;

  for (int tile = sub; tile < TILES_PER_B; tile += SUBWAVES) {
    // Gather pointer for this lane's row (redirect to updated rows via map).
    const int k0 = tile * 16 + m;
    int ridx = 0;
    if (k0 < KP1) ridx = (k0 == 0) ? pos_ridx : crow[k0 - 1];
    const int mj = map[ridx];
    const float* rp = (mj >= 0) ? (upd + mj * FD) : (mem + (size_t)ridx * FD);

    v8f acc = {0.f, 0.f, 0.f, 0.f, 0.f, 0.f, 0.f, 0.f};
#pragma unroll
    for (int c = 0; c < 4; ++c) {
      float ta[16] __attribute__((aligned(16)));
      const float4* p0 = (const float4*)(rp + 32 * c + 8 * h);
      const float4* p1 = (const float4*)(rp + 32 * c + 16 + 8 * h);
      *(float4*)(ta + 0)  = p0[0];
      *(float4*)(ta + 4)  = p0[1];
      *(float4*)(ta + 8)  = p1[0];
      *(float4*)(ta + 12) = p1[1];
      v16h av;
#pragma unroll
      for (int q = 0; q < 16; ++q) av[q] = (_Float16)ta[q];
      acc = __builtin_amdgcn_wmma_f32_16x16x32_f16(
          /*neg_a=*/false, av, /*neg_b=*/false, bv[c],
          /*c_mod=*/(short)0, acc, /*reuse_a=*/false, /*reuse_b=*/false);
    }

    // Lane 0 holds rows 0..7 of D in acc[0..7] (col 0); lane 16 rows 8..15.
    if (m == 0) {
      const int kb = tile * 16 + 8 * h;
#pragma unroll
      for (int v = 0; v < 8; ++v) {
        const int kk = kb + v;
        if (kk < KP1) {
          const float ev = __expf(acc[v] * INV_T);
          rawp[kk] = ev;
          zs += ev;
        }
      }
    }
  }

  // Two slots per wave; waves 0..2047 are dir 0, 2048..4095 are dir 1.
  if (m == 0) zslot[wid * 2 + h] = zs;
}

// ---------------------------------------------------------------------------
// Z1/Z2 = mean(exp) * N_DATA, from the fixed per-wave partial slots.
// Slots [0, 4096) belong to dir 0, [4096, 8192) to dir 1.
// ---------------------------------------------------------------------------
__global__ void __launch_bounds__(256) k_zred(const float* __restrict__ zslot,
                                              float* __restrict__ zvals) {
  __shared__ float red[256];
  const int t = threadIdx.x;
  float s0 = 0.f, s1 = 0.f;
  for (int i = t; i < TOTAL_WAVES; i += 256) {
    s0 += zslot[i];
    s1 += zslot[TOTAL_WAVES + i];
  }
  red[t] = s0; __syncthreads();
  for (int s = 128; s > 0; s >>= 1) { if (t < s) red[t] += red[t + s]; __syncthreads(); }
  if (t == 0) zvals[0] = red[0] * ((float)NDATA / (float)(B_SZ * KP1));
  __syncthreads();
  red[t] = s1; __syncthreads();
  for (int s = 128; s > 0; s >>= 1) { if (t < s) red[t] += red[t + s]; __syncthreads(); }
  if (t == 0) zvals[1] = red[0] * ((float)NDATA / (float)(B_SZ * KP1));
}

// ---------------------------------------------------------------------------
// Loss terms: k==0 -> -log(P/(P+r+eps)); k>0 -> -log(r/(P+r+eps)); P=raw/Z.
// 512 fixed blocks, deterministic tree reduce to per-block partials.
// ---------------------------------------------------------------------------
__global__ void __launch_bounds__(256) k_losspart(
    const float* __restrict__ raw, const float* __restrict__ zvals,
    float* __restrict__ lpart)
{
  __shared__ float red[256];
  const float Z1 = zvals[0], Z2 = zvals[1];
  float s = 0.f;
  const int NTOT = 2 * NPE;
  for (int e = blockIdx.x * blockDim.x + threadIdx.x; e < NTOT;
       e += gridDim.x * blockDim.x) {
    const int d1 = (e >= NPE) ? 1 : 0;
    const int r  = e - d1 * NPE;
    const int k  = r % KP1;
    const float P   = raw[e] / (d1 ? Z2 : Z1);
    const float den = P + RES_C + EPS_C;
    s -= (k == 0) ? __logf(P / den) : __logf(RES_C / den);
  }
  const int t = threadIdx.x;
  red[t] = s; __syncthreads();
  for (int st = 128; st > 0; st >>= 1) { if (t < st) red[t] += red[t + st]; __syncthreads(); }
  if (t == 0) lpart[blockIdx.x] = red[0];
}

__global__ void __launch_bounds__(256) k_final(const float* __restrict__ lpart,
                                               float* __restrict__ out) {
  __shared__ float red[256];
  const int t = threadIdx.x;
  red[t] = lpart[t] + lpart[t + 256];   // 512 partials
  __syncthreads();
  for (int s = 128; s > 0; s >>= 1) { if (t < s) red[t] += red[t + s]; __syncthreads(); }
  if (t == 0) out[0] = red[0] / (float)B_SZ;   // mean over batch, both losses summed
}

// ---------------------------------------------------------------------------
extern "C" void kernel_launch(void* const* d_in, const int* in_sizes, int n_in,
                              void* d_out, int out_size, void* d_ws, size_t ws_size,
                              hipStream_t stream) {
  (void)in_sizes; (void)n_in; (void)out_size; (void)ws_size;
  const float* f_s  = (const float*)d_in[0];
  const float* f_t  = (const float*)d_in[1];
  const float* W_s  = (const float*)d_in[2];
  const float* b_s  = (const float*)d_in[3];
  const float* W_t  = (const float*)d_in[4];
  const float* b_t  = (const float*)d_in[5];
  const float* mem1 = (const float*)d_in[6];
  const float* mem2 = (const float*)d_in[7];
  const int*   idx  = (const int*)d_in[8];
  const int*   cidx = (const int*)d_in[9];

  // Workspace layout (floats); total ~13 MB.
  float* ws    = (float*)d_ws;
  int*   map   = (int*)ws;                       // NDATA ints
  float* es    = ws + NDATA;                     // 256*128
  float* et    = es + B_SZ * FD;
  float* upd1  = et + B_SZ * FD;
  float* upd2  = upd1 + B_SZ * FD;
  float* raw   = upd2 + B_SZ * FD;               // 2*NPE exp values
  float* zslot = raw + 2 * NPE;                  // 2*TOTAL_WAVES
  float* zvals = zslot + 2 * TOTAL_WAVES;        // Z1, Z2
  float* lpart = zvals + 2;                      // 512 loss partials

  k_init_map<<<512, 256, 0, stream>>>(map);
  k_scatter<<<1, 256, 0, stream>>>(idx, map);
  k_proj<<<dim3(B_SZ, 2), 128, 0, stream>>>(f_s, f_t, W_s, b_s, W_t, b_t,
                                            mem1, mem2, idx, es, et, upd1, upd2);
  k_main<<<MAIN_BLOCKS, MAIN_THREADS, 0, stream>>>(es, et, upd1, upd2,
                                                   mem1, mem2, idx, cidx, map,
                                                   raw, zslot);
  k_zred<<<1, 256, 0, stream>>>(zslot, zvals);
  k_losspart<<<512, 256, 0, stream>>>(raw, zvals, lpart);
  k_final<<<1, 256, 0, stream>>>(lpart, (float*)d_out);
}
